// RNN_6605659701957
// MI455X (gfx1250) — matrix operand order repaired and checked by reading, not compiled, not run on hardware
//
#include <hip/hip_runtime.h>
#include <hip/hip_bf16.h>

// Problem constants (from reference): SEQ=2048, B=64, D_IN=D_H=512
#define S_LEN 2048
#define BATCH 64
#define DDIM  512
#define M_TOTAL (S_LEN * BATCH)                // 131072 rows of the i2h GEMM
#define OUT_TAIL ((long)S_LEN * BATCH * DDIM)  // offset of h_last in d_out

typedef __attribute__((ext_vector_type(16))) _Float16 v16h;
typedef __attribute__((ext_vector_type(8)))  _Float16 v8h;
typedef __attribute__((ext_vector_type(4)))  _Float16 v4h;
typedef __attribute__((ext_vector_type(8)))  float    v8f;
typedef __attribute__((ext_vector_type(4)))  float    v4f;

// ---------------------------------------------------------------------------
// Fragment loaders for v_wmma_f32_16x16x32_f16 (wave32).
// A (16x32, MxK): lane holds row m = lane%16.
//   VGPR v: k = 16*(v>=4) + 8*half + 2*(v&3)  -> two contiguous 8-half runs:
//   [kc + 8*half .. +7] -> frag[0..7],  [kc + 16 + 8*half .. +7] -> frag[8..15]
// B (32x16, KxN): lane holds column n = lane%16.
//   same k order as A -> two contiguous 8-half runs:
//   [kc + 8*half .. +7] -> frag[0..7],  [kc + 16 + 8*half .. +7] -> frag[8..15]
// ---------------------------------------------------------------------------
__device__ inline v16h frag_a_f32(const float* __restrict__ p, int half) {
    const float* q1 = p + half * 8;
    const float* q2 = p + 16 + half * 8;
    v4f f0 = *(const v4f*)(q1);
    v4f f1 = *(const v4f*)(q1 + 4);
    v4f f2 = *(const v4f*)(q2);
    v4f f3 = *(const v4f*)(q2 + 4);
    v16h r;
#pragma unroll
    for (int i = 0; i < 4; ++i) {
        r[i]      = (_Float16)f0[i];
        r[4 + i]  = (_Float16)f1[i];
        r[8 + i]  = (_Float16)f2[i];
        r[12 + i] = (_Float16)f3[i];
    }
    return r;
}

__device__ inline v16h frag_a_f16(const _Float16* p, int half) {
    v8h lo = *(const v8h*)(p + half * 8);
    v8h hi = *(const v8h*)(p + 16 + half * 8);
    v16h r;
#pragma unroll
    for (int i = 0; i < 8; ++i) { r[i] = lo[i]; r[8 + i] = hi[i]; }
    return r;
}

__device__ inline v16h frag_b_f16(const _Float16* __restrict__ p, int half) {
    v8h lo = *(const v8h*)(p + half * 8);
    v8h hi = *(const v8h*)(p + 16 + half * 8);
    v16h r;
#pragma unroll
    for (int i = 0; i < 8; ++i) { r[i] = lo[i]; r[8 + i] = hi[i]; }
    return r;
}

__device__ inline v8f wmma_f16(v16h a, v16h b, v8f c) {
    return __builtin_amdgcn_wmma_f32_16x16x32_f16(false, a, false, b,
                                                  (short)0, c, false, false);
}

// ---------------------------------------------------------------------------
// Kernel 0: one-time fp32 -> f16 conversion of a weight matrix (512x512).
// ---------------------------------------------------------------------------
__global__ void convert_w_kernel(const float* __restrict__ w,
                                 _Float16* __restrict__ o) {
    int i = (blockIdx.x * blockDim.x + threadIdx.x) * 4;
    v4f f = *(const v4f*)(w + i);
    v4h h;
#pragma unroll
    for (int j = 0; j < 4; ++j) h[j] = (_Float16)f[j];
    *(v4h*)(o + i) = h;
}

// ---------------------------------------------------------------------------
// Kernel 1: xi = x @ W_i2h^T + b_i2h, written directly into d_out[0 .. S*B*D).
// Block tile 128(M) x 256(N), 256 threads = 8 waves arranged 4(M) x 2(N);
// each wave computes a 32x128 tile = 2x8 WMMA accumulators over K=512.
// W is pre-converted f16 (no cvt VALU for B); only the streamed x operand is
// converted on the fly (16 v_cvt_pk per lane per chunk vs 16 WMMAs).
// ---------------------------------------------------------------------------
__global__ __launch_bounds__(256) void i2h_gemm_kernel(
    const float* __restrict__ x, const _Float16* __restrict__ W,
    const float* __restrict__ bias, float* __restrict__ out) {
    const int lane = threadIdx.x & 31;
    const int wave = threadIdx.x >> 5;   // 0..7
    const int half = lane >> 4;
    const int lr   = lane & 15;
    const int wm   = wave & 3;           // 4 waves along M
    const int wn   = wave >> 2;          // 2 waves along N
    const long Moff = (long)blockIdx.x * 128 + wm * 32;
    const int  Noff = blockIdx.y * 256 + wn * 128;

    v8f acc[2][8];
#pragma unroll
    for (int mt = 0; mt < 2; ++mt)
#pragma unroll
        for (int nt = 0; nt < 8; ++nt) acc[mt][nt] = (v8f){};

    const float* arow0 = x + (Moff + 0 * 16 + lr) * (long)DDIM;
    const float* arow1 = x + (Moff + 1 * 16 + lr) * (long)DDIM;
    const _Float16* brow[8];
#pragma unroll
    for (int nt = 0; nt < 8; ++nt)
        brow[nt] = W + (long)(Noff + nt * 16 + lr) * DDIM;

    for (int kc = 0; kc < DDIM; kc += 32) {
        v16h a0 = frag_a_f32(arow0 + kc, half);
        v16h a1 = frag_a_f32(arow1 + kc, half);
#pragma unroll
        for (int nt = 0; nt < 8; ++nt) {
            v16h b = frag_b_f16(brow[nt] + kc, half);
            acc[0][nt] = wmma_f16(a0, b, acc[0][nt]);
            acc[1][nt] = wmma_f16(a1, b, acc[1][nt]);
        }
    }

#pragma unroll
    for (int nt = 0; nt < 8; ++nt) {
        const int   n  = Noff + nt * 16 + lr;
        const float bv = bias[n];
#pragma unroll
        for (int mt = 0; mt < 2; ++mt) {
#pragma unroll
            for (int v = 0; v < 8; ++v) {
                long m = Moff + mt * 16 + v + 8 * half;
                out[m * DDIM + n] = acc[mt][nt][v] + bv;
            }
        }
    }
}

// ---------------------------------------------------------------------------
// Kernel 2: sequential scan. Batch rows are independent -> 4 workgroups of
// 1024 threads (32 waves, one full WGP) each own 16 batch rows; the only
// per-step sync is the workgroup barrier. h is kept in LDS as f16 with a
// padded row stride (520 halves) so ds_load_b128 A-fragment reads spread
// across banks. Each wave owns one 16-column tile.
//
// Key optimization: each wave's 16 W_h2h B-fragments (16 rows x K=512 f16 =
// 128 VGPRs) are invariant across all 2048 timesteps, so they are hoisted
// into registers before the t-loop. The steady-state step is then just
// 32x ds_load_b128 + 16x v_wmma_f32_16x16x32_f16 + tanh epilogue — zero
// global loads on the serial critical path.
// ---------------------------------------------------------------------------
#define HROW 520  // LDS row stride in halves (padding avoids bank conflicts)

__global__ __launch_bounds__(1024) void rnn_scan_kernel(
    const _Float16* __restrict__ Wh,  // W_h2h as f16, row-major [512][512]
    const float* __restrict__ bh, const float* __restrict__ h0,
    float* __restrict__ out) {
    __shared__ _Float16 hsm[16 * HROW];

    const int tid  = threadIdx.x;
    const int lane = tid & 31;
    const int wave = tid >> 5;          // 0..31 -> 16-wide column tile
    const int half = lane >> 4;
    const int lr   = lane & 15;
    const int bg   = blockIdx.x * 16;   // first batch row owned by this block

    // Load h0 slice into LDS as f16.
    for (int idx = tid; idx < 16 * DDIM; idx += 1024) {
        int r = idx >> 9, c = idx & (DDIM - 1);
        hsm[r * HROW + c] = (_Float16)h0[(bg + r) * DDIM + c];
    }
    __syncthreads();

    const int n = wave * 16 + lr;             // output column of this lane
    const float bhn = bh[n];
    const _Float16* wrow = Wh + (long)n * DDIM;

    // Hoist the time-invariant W_h2h B-fragments into registers (128 VGPRs).
    v16h wfrag[16];
#pragma unroll
    for (int c = 0; c < 16; ++c) wfrag[c] = frag_b_f16(wrow + c * 32, half);

    for (int t = 0; t < S_LEN; ++t) {
        v8f acc = (v8f){};
#pragma unroll
        for (int c = 0; c < 16; ++c) {
            v16h a = frag_a_f16(&hsm[lr * HROW + c * 32], half);
            acc = wmma_f16(a, wfrag[c], acc);
        }

        // h_new = tanh(acc + xi_t + b); xi_t lives in out[t] and is
        // overwritten in place by h_t.
        float vals[8];
        const long rb = (long)t * BATCH + bg;
#pragma unroll
        for (int v = 0; v < 8; ++v) {
            int  m = v + 8 * half;
            long g = (rb + m) * DDIM + n;
            float hv = tanhf(acc[v] + out[g] + bhn);
            out[g]  = hv;
            vals[v] = hv;
            if (t == S_LEN - 1)
                out[OUT_TAIL + (long)(bg + m) * DDIM + n] = hv;  // h_last
        }

        __syncthreads();  // everyone finished reading old h from LDS
#pragma unroll
        for (int v = 0; v < 8; ++v) {
            int m = v + 8 * half;
            hsm[m * HROW + n] = (_Float16)vals[v];
        }
        __syncthreads();  // new h visible for next step
    }
}

// ---------------------------------------------------------------------------
extern "C" void kernel_launch(void* const* d_in, const int* in_sizes, int n_in,
                              void* d_out, int out_size, void* d_ws,
                              size_t ws_size, hipStream_t stream) {
    const float* x     = (const float*)d_in[0];  // (S, B, D_in)
    const float* h0    = (const float*)d_in[1];  // (B, D_h)
    const float* W_i2h = (const float*)d_in[2];  // (D_h, D_in)
    const float* b_i2h = (const float*)d_in[3];  // (D_h,)
    const float* W_h2h = (const float*)d_in[4];  // (D_h, D_h)
    const float* b_h2h = (const float*)d_in[5];  // (D_h,)
    float* out = (float*)d_out;                  // outputs | h_last

    _Float16* Wi_f16 = (_Float16*)d_ws;                  // 512 KB
    _Float16* Wh_f16 = (_Float16*)d_ws + DDIM * DDIM;    // 512 KB

    // Kernel 0 (x2): weights fp32 -> f16 (262144 elements each, 4/thread).
    convert_w_kernel<<<dim3(DDIM * DDIM / (256 * 4)), dim3(256), 0, stream>>>(
        W_i2h, Wi_f16);
    convert_w_kernel<<<dim3(DDIM * DDIM / (256 * 4)), dim3(256), 0, stream>>>(
        W_h2h, Wh_f16);

    // Kernel 1: xi for all timesteps into d_out. Grid: 1024 M-tiles x 2 N-tiles.
    i2h_gemm_kernel<<<dim3(M_TOTAL / 128, DDIM / 256), dim3(256), 0, stream>>>(
        x, Wi_f16, b_i2h, out);

    // Kernel 2: sequential recurrence, 4 independent batch groups.
    rnn_scan_kernel<<<dim3(BATCH / 16), dim3(1024), 0, stream>>>(
        Wh_f16, b_h2h, h0, out);
}
